// _FramewiseGeometryInteraction_71957882077197
// MI455X (gfx1250) — matrix operand order repaired and checked by reading, not compiled, not run on hardware
//
#include <hip/hip_runtime.h>

#define L_TOK 2304
#define HID   2048
#define NH    16
#define HD    128
#define SPLITH 8

typedef __bf16 bf16;
typedef __attribute__((ext_vector_type(16))) __bf16 bf16x16;
typedef __attribute__((ext_vector_type(8)))  __bf16 bf16x8;
typedef __attribute__((ext_vector_type(8)))  float  f32x8;

#define WMMA_BF16(a, b, c) \
  __builtin_amdgcn_wmma_f32_16x16x32_bf16(false, (a), false, (b), (short)0, (c), false, false)

static __device__ __forceinline__ f32x8 zero_f32x8() {
  f32x8 z;
#pragma unroll
  for (int i = 0; i < 8; ++i) z[i] = 0.0f;
  return z;
}

// A-matrix fragment, 16x32 bf16, row-major source.
// Lane r (r=lane&15) holds row M=r; VGPR0-3: K = half*8 + 0..7,
// VGPR4-7: K = 16 + half*8 + 0..7  ->  two contiguous 16B chunks per lane.
static __device__ __forceinline__ bf16x16
load_afrag(const bf16* __restrict__ base, int row, int ld, int kBase, int half) {
  const bf16x8* p0 = reinterpret_cast<const bf16x8*>(base + (size_t)row * ld + kBase + half * 8);
  const bf16x8* p1 = reinterpret_cast<const bf16x8*>(base + (size_t)row * ld + kBase + 16 + half * 8);
  bf16x8 a = *p0, b = *p1;
  bf16x16 r;
#pragma unroll
  for (int i = 0; i < 8; ++i) { r[i] = a[i]; r[i + 8] = b[i]; }
  return r;
}

// B-matrix fragment, 32x16 bf16, where B[k][n] = src[n][k] (src row-major).
// Lane holds column N=lane&15; K = half*8 + 0..7, then 16 + half*8 + 0..7 -> two 16B chunks.
static __device__ __forceinline__ bf16x16
load_bfrag(const bf16* __restrict__ base, int nrow, int ld, int kBase, int half) {
  const bf16x8* p0 = reinterpret_cast<const bf16x8*>(base + (size_t)nrow * ld + kBase + half * 8);
  const bf16x8* p1 = reinterpret_cast<const bf16x8*>(base + (size_t)nrow * ld + kBase + 16 + half * 8);
  bf16x8 a = *p0, b = *p1;
  bf16x16 r;
#pragma unroll
  for (int i = 0; i < 8; ++i) { r[i] = a[i]; r[i + 8] = b[i]; }
  return r;
}

// ---------------------------------------------------------------------------
// fp32 -> bf16 elementwise conversion
// ---------------------------------------------------------------------------
__global__ void cvt_bf16_kernel(const float* __restrict__ src, bf16* __restrict__ dst, int n) {
  int idx = blockIdx.x * blockDim.x + threadIdx.x;
  int stride = gridDim.x * blockDim.x;
  for (int i = idx; i < n; i += stride) dst[i] = (bf16)src[i];
}

// ---------------------------------------------------------------------------
// LayerNorm over rows of [L_TOK, HID]; bf16 output
// ---------------------------------------------------------------------------
__global__ void __launch_bounds__(256)
ln_kernel(const float* __restrict__ x, const float* __restrict__ w,
          const float* __restrict__ b, bf16* __restrict__ out) {
  const int row = blockIdx.x;
  const int t = threadIdx.x;
  const float* xr = x + (size_t)row * HID;
  float v[8];
  float s = 0.f, s2 = 0.f;
#pragma unroll
  for (int i = 0; i < 8; ++i) { v[i] = xr[t + i * 256]; s += v[i]; s2 += v[i] * v[i]; }
#pragma unroll
  for (int o = 16; o >= 1; o >>= 1) { s += __shfl_xor(s, o, 32); s2 += __shfl_xor(s2, o, 32); }
  __shared__ float red[2][8];
  const int wv = t >> 5;
  if ((t & 31) == 0) { red[0][wv] = s; red[1][wv] = s2; }
  __syncthreads();
  if (t == 0) {
    float a = 0.f, c = 0.f;
#pragma unroll
    for (int i = 0; i < 8; ++i) { a += red[0][i]; c += red[1][i]; }
    red[0][0] = a; red[1][0] = c;
  }
  __syncthreads();
  const float mu = red[0][0] * (1.0f / HID);
  const float var = red[1][0] * (1.0f / HID) - mu * mu;
  const float rs = rsqrtf(var + 1e-6f);
#pragma unroll
  for (int i = 0; i < 8; ++i) {
    int c = t + i * 256;
    out[(size_t)row * HID + c] = (bf16)((v[i] - mu) * rs * w[c] + b[c]);
  }
}

// ---------------------------------------------------------------------------
// WMMA GEMM: Out[M,N] = A[M,K] * W[N,K]^T  (torch Linear convention)
// One wave computes a 32x64 output tile; register double-buffered over K so
// next-k fragments load while current WMMAs execute. Requires M%32==0, K%64==0.
// mode 0: bf16 row-major out       mode 1: bf16 transposed out (Out[n][m])
// mode 2: f32 row-major * tanh(*scale)   mode 3: bf16 relu(acc + bias[n])
// ---------------------------------------------------------------------------
__global__ void __launch_bounds__(128)
gemm_bf16_wmma(const bf16* __restrict__ A, const bf16* __restrict__ W,
               void* __restrict__ Out, const float* __restrict__ bias,
               const float* __restrict__ scale_ptr,
               int M, int N, int K, int mode) {
  const int wid  = (blockIdx.x * blockDim.x + threadIdx.x) >> 5;
  const int lane = threadIdx.x & 31;
  const int half = lane >> 4;
  const int l15  = lane & 15;
  const int tilesN = N >> 6;
  const int nTiles = (M >> 5) * tilesN;
  if (wid >= nTiles) return;
  const int m0 = (wid / tilesN) << 5;
  const int n0 = (wid % tilesN) << 6;

  f32x8 acc0[4], acc1[4];
#pragma unroll
  for (int t = 0; t < 4; ++t) { acc0[t] = zero_f32x8(); acc1[t] = zero_f32x8(); }

  // double-buffered fragment sets
  bf16x16 aA[2], aB[2], bB[2][4];
  aA[0] = load_afrag(A, m0 + l15, K, 0, half);
  aB[0] = load_afrag(A, m0 + 16 + l15, K, 0, half);
#pragma unroll
  for (int t = 0; t < 4; ++t) bB[0][t] = load_bfrag(W, n0 + t * 16 + l15, K, 0, half);

#pragma unroll 1
  for (int kb = 0; kb < K; kb += 64) {
    // issue loads for set 1 (kb+32), always in range since K%64==0
    aA[1] = load_afrag(A, m0 + l15, K, kb + 32, half);
    aB[1] = load_afrag(A, m0 + 16 + l15, K, kb + 32, half);
#pragma unroll
    for (int t = 0; t < 4; ++t) bB[1][t] = load_bfrag(W, n0 + t * 16 + l15, K, kb + 32, half);

    // WMMAs on set 0 (overlaps with set-1 loads)
#pragma unroll
    for (int t = 0; t < 4; ++t) {
      acc0[t] = WMMA_BF16(aA[0], bB[0][t], acc0[t]);
      acc1[t] = WMMA_BF16(aB[0], bB[0][t], acc1[t]);
    }

    // issue loads for set 0 (kb+64) if in range
    if (kb + 64 < K) {
      aA[0] = load_afrag(A, m0 + l15, K, kb + 64, half);
      aB[0] = load_afrag(A, m0 + 16 + l15, K, kb + 64, half);
#pragma unroll
      for (int t = 0; t < 4; ++t) bB[0][t] = load_bfrag(W, n0 + t * 16 + l15, K, kb + 64, half);
    }

    // WMMAs on set 1
#pragma unroll
    for (int t = 0; t < 4; ++t) {
      acc0[t] = WMMA_BF16(aA[1], bB[1][t], acc0[t]);
      acc1[t] = WMMA_BF16(aB[1], bB[1][t], acc1[t]);
    }
  }

  const float scl = (mode == 2) ? tanhf(scale_ptr[0]) : 1.0f;
#pragma unroll
  for (int r = 0; r < 2; ++r) {
#pragma unroll
    for (int t = 0; t < 4; ++t) {
      const int n = n0 + t * 16 + l15;
#pragma unroll
      for (int i = 0; i < 8; ++i) {
        const int m = m0 + r * 16 + half * 8 + i;
        float v = (r == 0) ? acc0[t][i] : acc1[t][i];
        if (mode == 3) { v += bias[n]; v = v > 0.0f ? v : 0.0f; }
        if (mode == 2)      ((float*)Out)[(size_t)m * N + n] = v * scl;
        else if (mode == 1) ((bf16*)Out)[(size_t)n * M + m] = (bf16)v;
        else                ((bf16*)Out)[(size_t)m * N + n] = (bf16)v;
      }
    }
  }
}

// ---------------------------------------------------------------------------
// Per-token scalars: logimp[l] = log(sigmoid(h1[l]·w2 + b2) + 0.1)
//                    gate[j][l] = sigmoid(ki[l]·bg_w[j] + bg_b[j]), j<8
// One wave per token.
// ---------------------------------------------------------------------------
__global__ void __launch_bounds__(256)
gateimp_kernel(const bf16* __restrict__ ki, const bf16* __restrict__ h1,
               const float* __restrict__ w2, const float* __restrict__ b2,
               const float* __restrict__ bgw, const float* __restrict__ bgb,
               float* __restrict__ logimp, float* __restrict__ gate) {
  const int wid  = (blockIdx.x * blockDim.x + threadIdx.x) >> 5;
  const int lane = threadIdx.x & 31;
  if (wid >= L_TOK) return;

  float acc = 0.f;
  for (int k = lane; k < (HID / 4); k += 32)
    acc += (float)h1[(size_t)wid * (HID / 4) + k] * w2[k];

  float g[8];
#pragma unroll
  for (int j = 0; j < 8; ++j) g[j] = 0.f;
  for (int k = lane; k < HID; k += 32) {
    const float xv = (float)ki[(size_t)wid * HID + k];
#pragma unroll
    for (int j = 0; j < 8; ++j) g[j] += xv * bgw[j * HID + k];
  }
#pragma unroll
  for (int off = 16; off >= 1; off >>= 1) {
    acc += __shfl_xor(acc, off, 32);
#pragma unroll
    for (int j = 0; j < 8; ++j) g[j] += __shfl_xor(g[j], off, 32);
  }
  if (lane == 0) {
    const float imp = 1.0f / (1.0f + __expf(-(acc + b2[0])));
    logimp[wid] = logf(imp + 0.1f);
#pragma unroll
    for (int j = 0; j < 8; ++j)
      gate[j * L_TOK + wid] = 1.0f / (1.0f + __expf(-(g[j] + bgb[j])));
  }
}

// ---------------------------------------------------------------------------
// Flash attention. One wave = (head, 16 query rows). Computes S^T = K·Q^T so
// softmax columns == queries == one lane each, and exp(S^T) lands directly in
// the A-fragment element order for O += P·V (V pre-transposed: Vt[d][l]).
// K A-fragments and V B-fragments are software-pipelined one step ahead.
// ---------------------------------------------------------------------------
__global__ void __launch_bounds__(128)
attn_kernel(const bf16* __restrict__ Qm, const bf16* __restrict__ Km,
            const bf16* __restrict__ Vt, const float* __restrict__ logimp,
            const float* __restrict__ gatesc, bf16* __restrict__ AOut) {
  const int wid  = (blockIdx.x * blockDim.x + threadIdx.x) >> 5;
  const int lane = threadIdx.x & 31;
  const int half = lane >> 4;
  const int l15  = lane & 15;
  const int head = wid / (L_TOK / 16);
  const int q0   = (wid % (L_TOK / 16)) << 4;
  const int dh0  = head * HD;

  // Q^T B-fragments (persistent): B[k][n] = Q[q0+n][dh0+k]
  bf16x16 qb[4];
#pragma unroll
  for (int f = 0; f < 4; ++f) qb[f] = load_bfrag(Qm, q0 + l15, HID, dh0 + f * 32, half);

  const int q = q0 + l15;
  const float qy = (float)(q / 48), qx = (float)(q % 48);
  const float negInvDiag = -1.0f / (sqrtf(48.0f * 48.0f + 48.0f * 48.0f) + 1e-6f);
  const bool useSpatial = (head < SPLITH);
  const float gs = useSpatial ? gatesc[head * L_TOK + q] : 0.0f;
  const float sscale = 0.08838834764831845f; // 1/sqrt(128)

  float run_max = -1e30f, run_sum = 0.0f;
  f32x8 o[8];
#pragma unroll
  for (int t = 0; t < 8; ++t) o[t] = zero_f32x8();

  // prefetch first K A-fragments (f=0 of block 0)
  bf16x16 ak0 = load_afrag(Km, 0 + l15, HID, dh0, half);
  bf16x16 ak1 = load_afrag(Km, 16 + l15, HID, dh0, half);

#pragma unroll 1
  for (int k0 = 0; k0 < L_TOK; k0 += 32) {
    // S^T tiles: rows = keys, cols = queries; prefetch next f (or next block f=0)
    f32x8 s0 = zero_f32x8(), s1 = zero_f32x8();
#pragma unroll
    for (int f = 0; f < 4; ++f) {
      bf16x16 ca0 = ak0, ca1 = ak1;
      const int nf = (f + 1) & 3;
      const int nk = (f == 3) ? (k0 + 32) : k0;
      if (!(f == 3 && k0 + 32 >= L_TOK)) {
        ak0 = load_afrag(Km, nk + l15, HID, dh0 + nf * 32, half);
        ak1 = load_afrag(Km, nk + 16 + l15, HID, dh0 + nf * 32, half);
      }
      s0 = WMMA_BF16(ca0, qb[f], s0);
      s1 = WMMA_BF16(ca1, qb[f], s1);
    }

    // first V fragment in flight during the softmax VALU work below
    bf16x16 vb = load_bfrag(Vt, dh0 + l15, L_TOK, k0, half);

    // bias: lane holds query q, 16 key entries (8 per half-tile)
    float vals[16];
#pragma unroll
    for (int i = 0; i < 8; ++i) {
      const int key0 = k0 + half * 8 + i;
      const int key1 = key0 + 16;
      float b0 = logimp[key0], b1 = logimp[key1];
      if (useSpatial) {
        const float ky0 = (float)(key0 / 48), kx0 = (float)(key0 % 48);
        const float ky1 = (float)(key1 / 48), kx1 = (float)(key1 % 48);
        const float dy0 = qy - ky0, dx0 = qx - kx0;
        const float dy1 = qy - ky1, dx1 = qx - kx1;
        b0 += sqrtf(dy0 * dy0 + dx0 * dx0) * negInvDiag * gs;
        b1 += sqrtf(dy1 * dy1 + dx1 * dx1) * negInvDiag * gs;
      }
      vals[i]     = s0[i] * sscale + b0;
      vals[i + 8] = s1[i] * sscale + b1;
    }

    // online softmax over the query's 32 keys (16 local + other half)
    float mloc = vals[0];
#pragma unroll
    for (int j = 1; j < 16; ++j) mloc = fmaxf(mloc, vals[j]);
    mloc = fmaxf(mloc, __shfl_xor(mloc, 16, 32));
    const float mnew = fmaxf(run_max, mloc);
    const float corr = __expf(run_max - mnew);
    float ssum = 0.f;
#pragma unroll
    for (int j = 0; j < 16; ++j) { vals[j] = __expf(vals[j] - mnew); ssum += vals[j]; }
    ssum += __shfl_xor(ssum, 16, 32);
    run_sum = run_sum * corr + ssum;
    run_max = mnew;

    // rescale O rows (row m = half*8+i is query half*8+i; its corr lives on lane m)
#pragma unroll
    for (int i = 0; i < 8; ++i) {
      const float ci = __shfl(corr, (half << 3) + i, 32);
#pragma unroll
      for (int t = 0; t < 8; ++t) o[t][i] *= ci;
    }

    // P as A-fragment: element j needs key (v>=4?16:0)+half*8+2*(v&3)+(j&1)
    bf16x16 pa;
#pragma unroll
    for (int j = 0; j < 16; ++j) {
      const int v = j >> 1;
      const int gidx = ((v & 3) << 1) | (j & 1);
      pa[j] = (bf16)((v < 4) ? vals[gidx] : vals[8 + gidx]);
    }

    // O += P·V ; B[k][n] = Vt[dh0+t*16+n][k0+k]; pipeline V loads one t ahead
#pragma unroll
    for (int t = 0; t < 8; ++t) {
      bf16x16 cv = vb;
      if (t < 7) vb = load_bfrag(Vt, dh0 + (t + 1) * 16 + l15, L_TOK, k0, half);
      o[t] = WMMA_BF16(pa, cv, o[t]);
    }
  }

  float inv[8];
#pragma unroll
  for (int i = 0; i < 8; ++i) inv[i] = 1.0f / __shfl(run_sum, (half << 3) + i, 32);
#pragma unroll
  for (int t = 0; t < 8; ++t) {
#pragma unroll
    for (int i = 0; i < 8; ++i) {
      const int m = q0 + half * 8 + i;
      AOut[(size_t)m * HID + dh0 + t * 16 + l15] = (bf16)(o[t][i] * inv[i]);
    }
  }
}

// ---------------------------------------------------------------------------
extern "C" void kernel_launch(void* const* d_in, const int* in_sizes, int n_in,
                              void* d_out, int out_size, void* d_ws, size_t ws_size,
                              hipStream_t stream) {
  const float* q_tokens  = (const float*)d_in[0];
  const float* kv_tokens = (const float*)d_in[1];
  const float* ln_q_w    = (const float*)d_in[2];
  const float* ln_q_b    = (const float*)d_in[3];
  const float* ln_k_w    = (const float*)d_in[4];
  const float* ln_k_b    = (const float*)d_in[5];
  const float* Wq        = (const float*)d_in[6];
  const float* Wk        = (const float*)d_in[7];
  const float* Wv        = (const float*)d_in[8];
  const float* Wo        = (const float*)d_in[9];
  const float* gate_sc   = (const float*)d_in[10];
  const float* imp_w1    = (const float*)d_in[11];
  const float* imp_b1    = (const float*)d_in[12];
  const float* imp_w2    = (const float*)d_in[13];
  const float* imp_b2    = (const float*)d_in[14];
  const float* bg_w      = (const float*)d_in[15];
  const float* bg_b      = (const float*)d_in[16];

  char* ws = (char*)d_ws;
  size_t off = 0;
  auto take = [&](size_t bytes) { char* p = ws + off; off += (bytes + 255) & ~(size_t)255; return p; };

  bf16* Wq_bf = (bf16*)take((size_t)HID * HID * 2);
  bf16* Wk_bf = (bf16*)take((size_t)HID * HID * 2);
  bf16* Wv_bf = (bf16*)take((size_t)HID * HID * 2);
  bf16* Wo_bf = (bf16*)take((size_t)HID * HID * 2);
  bf16* W1_bf = (bf16*)take((size_t)(HID / 4) * HID * 2);
  bf16* qi_bf = (bf16*)take((size_t)L_TOK * HID * 2);
  bf16* ki_bf = (bf16*)take((size_t)L_TOK * HID * 2);
  bf16* Qm    = (bf16*)take((size_t)L_TOK * HID * 2);
  bf16* Km    = (bf16*)take((size_t)L_TOK * HID * 2);
  bf16* VtT   = (bf16*)take((size_t)HID * L_TOK * 2);   // transposed V [d][l]
  bf16* H1    = (bf16*)take((size_t)L_TOK * (HID / 4) * 2);
  bf16* AOut  = (bf16*)take((size_t)L_TOK * HID * 2);
  float* logimp = (float*)take((size_t)L_TOK * 4);
  float* gatesc = (float*)take((size_t)SPLITH * L_TOK * 4);

  // 1) fp32 -> bf16 weight conversions
  cvt_bf16_kernel<<<2048, 256, 0, stream>>>(Wq, Wq_bf, HID * HID);
  cvt_bf16_kernel<<<2048, 256, 0, stream>>>(Wk, Wk_bf, HID * HID);
  cvt_bf16_kernel<<<2048, 256, 0, stream>>>(Wv, Wv_bf, HID * HID);
  cvt_bf16_kernel<<<2048, 256, 0, stream>>>(Wo, Wo_bf, HID * HID);
  cvt_bf16_kernel<<<1024, 256, 0, stream>>>(imp_w1, W1_bf, (HID / 4) * HID);

  // 2) LayerNorms
  ln_kernel<<<L_TOK, 256, 0, stream>>>(q_tokens, ln_q_w, ln_q_b, qi_bf);
  ln_kernel<<<L_TOK, 256, 0, stream>>>(kv_tokens, ln_k_w, ln_k_b, ki_bf);

  // 3) Projections (WMMA): Q, K row-major; V transposed for attention B-frags
  const int blkQKV = ((L_TOK / 32) * (HID / 64)) / 4;      // 576
  gemm_bf16_wmma<<<blkQKV, 128, 0, stream>>>(qi_bf, Wq_bf, Qm,  nullptr, nullptr, L_TOK, HID, HID, 0);
  gemm_bf16_wmma<<<blkQKV, 128, 0, stream>>>(ki_bf, Wk_bf, Km,  nullptr, nullptr, L_TOK, HID, HID, 0);
  gemm_bf16_wmma<<<blkQKV, 128, 0, stream>>>(ki_bf, Wv_bf, VtT, nullptr, nullptr, L_TOK, HID, HID, 1);

  // 4) Importance MLP layer 1 (WMMA, bias+relu fused)
  const int blkH1 = ((L_TOK / 32) * ((HID / 4) / 64)) / 4; // 144
  gemm_bf16_wmma<<<blkH1, 128, 0, stream>>>(ki_bf, W1_bf, H1, imp_b1, nullptr, L_TOK, HID / 4, HID, 3);

  // 5) log-importance + spatial gate sigmoids (one wave per token)
  gateimp_kernel<<<L_TOK / 8, 256, 0, stream>>>(ki_bf, H1, imp_w2, imp_b2, bg_w, bg_b, logimp, gatesc);

  // 6) Flash attention (one wave per head x 16-query tile)
  attn_kernel<<<(NH * (L_TOK / 16)) / 4, 128, 0, stream>>>(Qm, Km, VtT, logimp, gatesc, AOut);

  // 7) Output projection, scaled by tanh(gate), fp32 out
  gemm_bf16_wmma<<<blkQKV, 128, 0, stream>>>(AOut, Wo_bf, d_out, nullptr, gate_sc, L_TOK, HID, HID, 2);

  (void)in_sizes; (void)n_in; (void)out_size; (void)ws_size;
}
